// MAGNOEncoder_72816875536550
// MI455X (gfx1250) — matrix operand order repaired and checked by reading, not compiled, not run on hardware
//
#include <hip/hip_runtime.h>

// MAGNO encoder: B=2, N=100000, M=16384, S=3, E=262144, CD=2, CIN=3, COUT=32, HID=64
#define B_    2
#define N_    100000
#define M_    16384
#define S_    3
#define E_    262144
#define COUT_ 32
#define HID_  64
#define WPB   4   // waves per block (128 threads, wave32)

typedef __attribute__((ext_vector_type(8)))  _Float16 v8h;
typedef __attribute__((ext_vector_type(16))) _Float16 v16h;
typedef __attribute__((ext_vector_type(8)))  float    v8f;
typedef int v4i __attribute__((vector_size(16)));     // matches async-LDS builtin param
#define AS1 __attribute__((address_space(1)))
#define AS3 __attribute__((address_space(3)))

// ---- hardware tanh (gfx1250 v_tanh_f32) with branchless fallback ----
#if __has_builtin(__builtin_amdgcn_tanhf)
  #define TANHF(x) __builtin_amdgcn_tanhf(x)
#elif __has_builtin(__builtin_amdgcn_tanh_f32)
  #define TANHF(x) __builtin_amdgcn_tanh_f32(x)
#else
__device__ __forceinline__ float fast_tanh_(float y) {
  // tanh(y) = 1 - 2/(e^{2y}+1); __expf -> v_exp_f32, rcp -> v_rcp_f32 (branchless)
  float e = __expf(2.0f * y);
  return 1.0f - 2.0f * __builtin_amdgcn_rcpf(e + 1.0f);
}
  #define TANHF(x) fast_tanh_(x)
#endif

// ---- optional CDNA5 async global->LDS copy path ----
#ifndef USE_ASYNC_LDS
#if __has_builtin(__builtin_amdgcn_global_load_async_to_lds_b128)
#define USE_ASYNC_LDS 1
#else
#define USE_ASYNC_LDS 0
#endif
#endif

__device__ __forceinline__ float gelu_tanh(float x) {
  // jax.nn.gelu default (approximate=True, tanh form)
  const float c0 = 0.7978845608028654f;  // sqrt(2/pi)
  const float c1 = 0.044715f;
  float y = c0 * fmaf(c1 * x * x, x, x); // c0*(x + c1*x^3)
  return 0.5f * x * (1.0f + TANHF(y));
}

__device__ __forceinline__ v16h combine8(v8h lo, v8h hi) {
  return __builtin_shufflevector(lo, hi, 0,1,2,3,4,5,6,7,8,9,10,11,12,13,14,15);
}

// ---------------- lifting: pn = pndata @ W_lift.T + b_lift  [B,N,32] ----------------
__global__ __launch_bounds__(256) void lift_kernel(const float* __restrict__ pndata,
                                                   const float* __restrict__ Wl,
                                                   const float* __restrict__ bl,
                                                   float* __restrict__ pn) {
  int idx = blockIdx.x * 256 + threadIdx.x;
  if (idx >= B_ * N_ * COUT_) return;
  int c  = idx & 31;
  int bn = idx >> 5;
  const float* p = pndata + bn * 3;
  const float* w = Wl + c * 3;
  pn[idx] = fmaf(p[0], w[0], fmaf(p[1], w[1], fmaf(p[2], w[2], bl[c])));
}

// ---------------- scale weights: softmax(relu(ltc@Ws1.T+bs1)@Ws2.T+bs2) [M,S] --------
__global__ __launch_bounds__(256) void sw_kernel(const float* __restrict__ ltc,
                                                 const float* __restrict__ Ws1,
                                                 const float* __restrict__ bs1,
                                                 const float* __restrict__ Ws2,
                                                 const float* __restrict__ bs2,
                                                 float* __restrict__ sw) {
  int m = blockIdx.x * 256 + threadIdx.x;
  if (m >= M_) return;
  float t0 = ltc[m * 2 + 0], t1 = ltc[m * 2 + 1];
  float h[16];
#pragma unroll
  for (int j = 0; j < 16; ++j)
    h[j] = fmaxf(fmaf(t0, Ws1[j * 2], fmaf(t1, Ws1[j * 2 + 1], bs1[j])), 0.0f);
  float lg[S_];
  float mx = -1e30f;
#pragma unroll
  for (int s = 0; s < S_; ++s) {
    float acc = bs2[s];
#pragma unroll
    for (int j = 0; j < 16; ++j) acc = fmaf(h[j], Ws2[s * 16 + j], acc);
    lg[s] = acc;
    mx = fmaxf(mx, acc);
  }
  float den = 0.0f;
#pragma unroll
  for (int s = 0; s < S_; ++s) { lg[s] = __expf(lg[s] - mx); den += lg[s]; }
  float inv = __builtin_amdgcn_rcpf(den);
#pragma unroll
  for (int s = 0; s < S_; ++s) sw[m * S_ + s] = lg[s] * inv;
}

// ---------------- main edge kernel: per-edge MLP on WMMA + segment atomics -----------
__global__ __launch_bounds__(128) void edge_kernel(
    const float* __restrict__ x_coord, const float* __restrict__ ltc,
    const int* __restrict__ nbr_idx,   const int* __restrict__ row_idx,
    const float* __restrict__ W1, const float* __restrict__ b1,
    const float* __restrict__ W2, const float* __restrict__ b2,
    const float* __restrict__ W3, const float* __restrict__ b3,
    const float* __restrict__ pn, float* __restrict__ sums, float* __restrict__ cnt) {
  __shared__ float w1s[HID_ * 4];
  __shared__ float b1s[HID_], b2s[HID_], b3s[COUT_];
  __shared__ __align__(32) _Float16 w2s[HID_ * HID_];   // [n][k] row-major f16
  __shared__ __align__(32) _Float16 w3s[COUT_ * HID_];  // [n][k] row-major f16
  __shared__ __align__(32) _Float16 h1s[WPB][16 * HID_];
  __shared__ __align__(32) _Float16 h2s[WPB][16 * HID_];
  __shared__ __align__(16) float pns[WPB][16 * COUT_];
  __shared__ int rows16[WPB][16];
  __shared__ int nbrs16[WPB][16];

  const int tid = threadIdx.x;
  // stage MLP weights once per block (f32 -> f16 for WMMA layers)
  for (int i = tid; i < HID_ * 4; i += 128) w1s[i] = W1[i];
  for (int i = tid; i < HID_; i += 128) { b1s[i] = b1[i]; b2s[i] = b2[i]; }
  for (int i = tid; i < COUT_; i += 128) b3s[i] = b3[i];
  for (int i = tid; i < HID_ * HID_; i += 128) w2s[i] = (_Float16)W2[i];
  for (int i = tid; i < COUT_ * HID_; i += 128) w3s[i] = (_Float16)W3[i];
  __syncthreads();

  const int wave = tid >> 5;
  const int lane = tid & 31;
  const int m    = lane & 15;   // edge index within 16-edge tile / A-matrix row
  const int half = lane >> 4;
  const int ebase = (blockIdx.x * WPB + wave) * 16;   // E%16==0: tile never crosses b,s
  const int bb = ebase / (S_ * E_);
  const int ss = (ebase - bb * S_ * E_) / E_;
  const int eidx = ebase + m;

  const int nbr = nbr_idx[eidx];
  const int row = row_idx[eidx];
  if (half == 0) { rows16[wave][m] = row; nbrs16[wave][m] = nbr; }

  const float a0 = x_coord[(bb * N_ + nbr) * 2 + 0];
  const float a1 = x_coord[(bb * N_ + nbr) * 2 + 1];
  const float a2 = ltc[row * 2 + 0];
  const float a3 = ltc[row * 2 + 1];

  // --- layer 1 (K=4, VALU): lane halves split the 64 hidden units of edge m.
  //     pack pairs of f16 and store as b32 to halve DS traffic.
#pragma unroll
  for (int jj = 0; jj < 32; jj += 2) {
    int j = half * 32 + jj;
    const float* w = &w1s[j * 4];
    float h0 = fmaf(a0, w[0], fmaf(a1, w[1], fmaf(a2, w[2], fmaf(a3, w[3], b1s[j]))));
    float h1v = fmaf(a0, w[4], fmaf(a1, w[5], fmaf(a2, w[6], fmaf(a3, w[7], b1s[j + 1]))));
    union { _Float16 h[2]; unsigned u; } pk;
    pk.h[0] = (_Float16)gelu_tanh(h0);
    pk.h[1] = (_Float16)gelu_tanh(h1v);
    *(unsigned*)&h1s[wave][m * HID_ + j] = pk.u;
  }
  __syncthreads();

  // cooperative gather of lifted features for the 16 edges (2 KB/tile)
#if USE_ASYNC_LDS
  {
#pragma unroll
    for (int it = 0; it < 4; ++it) {
      int chunk = lane + 32 * it;                 // 128 x 16B chunks
      int ee = chunk >> 3, part = chunk & 7;      // 8 chunks per 32-float pn row
      const float* g = &pn[(bb * N_ + nbrs16[wave][ee]) * COUT_ + part * 4];
      __builtin_amdgcn_global_load_async_to_lds_b128(
          (AS1 v4i*)g, (AS3 v4i*)&pns[wave][chunk * 4], 0, 0);
    }
    asm volatile("s_wait_asynccnt 0" ::: "memory");
  }
#else
  for (int i = lane; i < 16 * COUT_; i += 32) {
    int ee = i >> 5, c = i & 31;
    pns[wave][i] = pn[(bb * N_ + nbrs16[wave][ee]) * COUT_ + c];
  }
#endif

  // --- A fragments (16x32 f16 layout: elems 0..7 = K koff..koff+7, 8..15 = K 16+koff..) ---
  const int koff = half * 8;
  const _Float16* h1p = h1s[wave] + m * HID_;
  v16h A0 = combine8(*(const v8h*)(h1p +  0 + koff), *(const v8h*)(h1p + 16 + koff));
  v16h A1 = combine8(*(const v8h*)(h1p + 32 + koff), *(const v8h*)(h1p + 48 + koff));

  const int nl = lane & 15;     // B/C/D column
  const int kb = half * 8;      // B-matrix K offset for this half-wave

  // --- layer 2: [16x64] @ W2^T[64x64], 4 N-tiles x 2 K-chunks of v_wmma_f32_16x16x32_f16
#pragma unroll
  for (int t = 0; t < 4; ++t) {
    const _Float16* wrow = w2s + (t * 16 + nl) * HID_ + kb;
    v16h B0 = combine8(*(const v8h*)(wrow +  0), *(const v8h*)(wrow + 16));
    v16h B1 = combine8(*(const v8h*)(wrow + 32), *(const v8h*)(wrow + 48));
    float bv = b2s[t * 16 + nl];
    v8f c;
#pragma unroll
    for (int r = 0; r < 8; ++r) c[r] = bv;
    c = __builtin_amdgcn_wmma_f32_16x16x32_f16(false, A0, false, B0, (short)0, c, false, false);
    c = __builtin_amdgcn_wmma_f32_16x16x32_f16(false, A1, false, B1, (short)0, c, false, false);
#pragma unroll
    for (int r = 0; r < 8; ++r) {
      int mr = r + 8 * half;  // D layout: VGPR r holds M=r (lanes<16) / M=8+r (lanes>=16)
      h2s[wave][mr * HID_ + t * 16 + nl] = (_Float16)gelu_tanh(c[r]);
    }
  }
  __syncthreads();

  // --- layer 3: [16x64] @ W3^T[64x32], 2 N-tiles x 2 K-chunks ---
  const _Float16* h2p = h2s[wave] + m * HID_;
  v16h A2 = combine8(*(const v8h*)(h2p +  0 + koff), *(const v8h*)(h2p + 16 + koff));
  v16h A3 = combine8(*(const v8h*)(h2p + 32 + koff), *(const v8h*)(h2p + 48 + koff));

  float* srow = sums + (bb * S_ + ss) * (M_ * COUT_);
#pragma unroll
  for (int t = 0; t < 2; ++t) {
    const _Float16* wrow = w3s + (t * 16 + nl) * HID_ + kb;
    v16h B0 = combine8(*(const v8h*)(wrow +  0), *(const v8h*)(wrow + 16));
    v16h B1 = combine8(*(const v8h*)(wrow + 32), *(const v8h*)(wrow + 48));
    float bv = b3s[t * 16 + nl];
    v8f c;
#pragma unroll
    for (int r = 0; r < 8; ++r) c[r] = bv;
    c = __builtin_amdgcn_wmma_f32_16x16x32_f16(false, A2, false, B0, (short)0, c, false, false);
    c = __builtin_amdgcn_wmma_f32_16x16x32_f16(false, A3, false, B1, (short)0, c, false, false);

    // rows are sorted -> run-merge the 8 rows this lane holds before hitting L2 atomics.
    // (branching is uniform within each 16-lane half: same row sequence per half)
    const int cc = t * 16 + nl;
    int   curRow = rows16[wave][8 * half];
    float acc    = c[0] * pns[wave][(8 * half) * COUT_ + cc];
    for (int r = 1; r < 8; ++r) {
      int mr = r + 8 * half;
      int rw = rows16[wave][mr];
      float v = c[r] * pns[wave][mr * COUT_ + cc];
      if (rw == curRow) {
        acc += v;
      } else {
        atomicAdd(&srow[curRow * COUT_ + cc], acc);
        curRow = rw;
        acc = v;
      }
    }
    atomicAdd(&srow[curRow * COUT_ + cc], acc);
  }

  // per-edge counts: ballot run-length so only run leaders issue one atomic
  if (half == 0) {
    int prev = __shfl_up(row, 1, 32);
    bool leader = (m == 0) || (row != prev);
    unsigned int lm = (unsigned int)__ballot(leader) & 0xFFFFu;
    if (leader) {
      unsigned int above = lm & ~((2u << m) - 1u);  // leaders strictly above lane m
      int next = above ? (__ffs(above) - 1) : 16;
      atomicAdd(&cnt[(bb * S_ + ss) * M_ + row], (float)(next - m));
    }
  }
}

// ---------------- finalize: mean per segment, softmax-weighted sum over scales -------
__global__ __launch_bounds__(256) void finalize_kernel(const float* __restrict__ sums,
                                                       const float* __restrict__ cnt,
                                                       const float* __restrict__ sw,
                                                       float* __restrict__ out) {
  int idx = blockIdx.x * 256 + threadIdx.x;
  if (idx >= B_ * M_ * COUT_) return;
  int c  = idx & 31;
  int bm = idx >> 5;
  int b  = bm / M_;
  int m  = bm - b * M_;
  float acc = 0.0f;
#pragma unroll
  for (int s = 0; s < S_; ++s) {
    int seg = (b * S_ + s) * M_ + m;
    float v = sums[seg * COUT_ + c] / fmaxf(cnt[seg], 1.0f);
    acc = fmaf(sw[m * S_ + s], v, acc);
  }
  out[idx] = acc;
}

extern "C" void kernel_launch(void* const* d_in, const int* in_sizes, int n_in,
                              void* d_out, int out_size, void* d_ws, size_t ws_size,
                              hipStream_t stream) {
  (void)in_sizes; (void)n_in; (void)out_size; (void)ws_size;
  const float* x_coord = (const float*)d_in[0];
  const float* pndata  = (const float*)d_in[1];
  const float* ltc     = (const float*)d_in[2];
  const int*   nbr_idx = (const int*)d_in[3];
  const int*   row_idx = (const int*)d_in[4];
  const float* W_lift  = (const float*)d_in[5];
  const float* b_lift  = (const float*)d_in[6];
  const float* W1  = (const float*)d_in[7];
  const float* b1  = (const float*)d_in[8];
  const float* W2  = (const float*)d_in[9];
  const float* b2  = (const float*)d_in[10];
  const float* W3  = (const float*)d_in[11];
  const float* b3  = (const float*)d_in[12];
  const float* Ws1 = (const float*)d_in[13];
  const float* bs1 = (const float*)d_in[14];
  const float* Ws2 = (const float*)d_in[15];
  const float* bs2 = (const float*)d_in[16];

  // workspace layout (bytes)
  const size_t SUMS_B = (size_t)B_ * S_ * M_ * COUT_ * 4;   // 12,582,912
  const size_t CNT_B  = (size_t)B_ * S_ * M_ * 4;           //    393,216
  const size_t PN_B   = (size_t)B_ * N_ * COUT_ * 4;        // 25,600,000
  char* ws = (char*)d_ws;
  float* sums = (float*)(ws);
  float* cnt  = (float*)(ws + SUMS_B);
  float* pn   = (float*)(ws + SUMS_B + CNT_B);
  float* sw   = (float*)(ws + SUMS_B + CNT_B + PN_B);

  // zero the accumulators (sums + cnt are adjacent)
  (void)hipMemsetAsync(sums, 0, SUMS_B + CNT_B, stream);

  lift_kernel<<<(B_ * N_ * COUT_ + 255) / 256, 256, 0, stream>>>(pndata, W_lift, b_lift, pn);
  sw_kernel<<<(M_ + 255) / 256, 256, 0, stream>>>(ltc, Ws1, bs1, Ws2, bs2, sw);

  // B*S*E/16 = 98304 16-edge tiles; 4 waves/block -> 24576 blocks (exact)
  const int tiles  = (B_ * S_ * E_) / 16;
  const int blocks = tiles / WPB;
  edge_kernel<<<blocks, 128, 0, stream>>>(x_coord, ltc, nbr_idx, row_idx,
                                          W1, b1, W2, b2, W3, b3, pn, sums, cnt);

  finalize_kernel<<<(B_ * M_ * COUT_ + 255) / 256, 256, 0, stream>>>(sums, cnt, sw, (float*)d_out);
}
